// SSIM_2851858284728
// MI455X (gfx1250) — compile-verified, matrix operand order checked by reading, not run on hardware
//
#include <hip/hip_runtime.h>

// CDNA5 / gfx1250: wave32, WMMA f16 16x16x32 with f32 accumulate.
typedef __attribute__((ext_vector_type(16))) _Float16 v16h;
typedef __attribute__((ext_vector_type(8)))  float    v8f;

#define WAVES_PER_BLOCK 4
#define RSTR 36              // LDS row stride in floats: 16B-aligned rows, conflict-free
#define HW 512
#define PLANE (HW * HW)

__device__ __forceinline__ int reflect512(int i) {
    i = (i < 0) ? -i : i;            // jnp.pad 'reflect': -1 -> 1
    i = (i > 511) ? (1022 - i) : i;  // 512 -> 510
    return i;
}

__global__ __launch_bounds__(WAVES_PER_BLOCK * 32)
void ssim_wmma_kernel(const float* __restrict__ img1,
                      const float* __restrict__ img2,
                      float* __restrict__ out)
{
    __shared__ float sm1[WAVES_PER_BLOCK][32 * RSTR];
    __shared__ float sm2[WAVES_PER_BLOCK][32 * RSTR];

    const int lane   = threadIdx.x & 31;
    const int wave   = threadIdx.x >> 5;
    const int tileId = blockIdx.x * WAVES_PER_BLOCK + wave;   // 49152 tiles total
    const int plane  = tileId >> 10;           // 16*3 = 48 planes
    const int t      = tileId & 1023;          // 32x32 tiles per plane
    const int r0     = (t >> 5) << 4;
    const int c0     = (t & 31) << 4;
    const size_t pbase = (size_t)plane * PLANE;

    float* s1 = sm1[wave];
    float* s2 = sm2[wave];

    // ---- Stage reflect-padded 32x32 halo region into LDS (per-wave private) ----
    const int gc = reflect512(c0 - 5 + lane);
    #pragma unroll 4
    for (int r = 0; r < 32; ++r) {
        const int gr = reflect512(r0 - 5 + r);
        const size_t gidx = pbase + (size_t)gr * HW + gc;
        s1[r * RSTR + lane] = img1[gidx];
        s2[r * RSTR + lane] = img2[gidx];
    }

    // ---- Gaussian taps g[d] = exp(-(d-5)^2/4.5)/sum, computed closed-form ----
    float gsum = 0.f;
    #pragma unroll
    for (int i = 0; i < 11; ++i) {
        const float d = (float)(i - 5);
        gsum += __expf(-(d * d) * (1.f / 4.5f));
    }
    const float ginv = 1.f / gsum;

    const int m  = lane & 15;                  // row (A) / col (B,D) index of this lane
    const int kb = (lane & 16) ? 8 : 0;        // K-group offset per 16-bit operand layout

    // Banded weight matrix: identical per-lane register serves as
    //  B in the horizontal pass (W[k][n]=g[k-n]) and A in the vertical pass (W[m][k]=g[k-m]).
    v16h W;
    #pragma unroll
    for (int e = 0; e < 16; ++e) {
        const int K = e + kb + ((e >= 8) ? 8 : 0);
        const int d = K - m;
        float wv = 0.f;
        if (d >= 0 && d <= 10) {
            const float u = (float)(d - 5);
            wv = __expf(-(u * u) * (1.f / 4.5f)) * ginv;
        }
        W[e] = (_Float16)wv;
    }

    // ---- 5 quantities: x, y, x*x, y*y, x*y; each = 2 horizontal + 1 vertical WMMA ----
    v8f S[5];
    #pragma unroll
    for (int q = 0; q < 5; ++q) {
        v8f Dh[2];
        #pragma unroll
        for (int h = 0; h < 2; ++h) {
            // A operand: region rows 16h..16h+15, cols per lane-K pattern (two runs of 8)
            const int ro = (16 * h + m) * RSTR + kb;   // 16B aligned
            const float4 x0 = *(const float4*)&s1[ro];
            const float4 x1 = *(const float4*)&s1[ro + 4];
            const float4 x2 = *(const float4*)&s1[ro + 16];
            const float4 x3 = *(const float4*)&s1[ro + 20];
            const float4 y0 = *(const float4*)&s2[ro];
            const float4 y1 = *(const float4*)&s2[ro + 4];
            const float4 y2 = *(const float4*)&s2[ro + 16];
            const float4 y3 = *(const float4*)&s2[ro + 20];
            const float ax[16] = {x0.x, x0.y, x0.z, x0.w, x1.x, x1.y, x1.z, x1.w,
                                  x2.x, x2.y, x2.z, x2.w, x3.x, x3.y, x3.z, x3.w};
            const float ay[16] = {y0.x, y0.y, y0.z, y0.w, y1.x, y1.y, y1.z, y1.w,
                                  y2.x, y2.y, y2.z, y2.w, y3.x, y3.y, y3.z, y3.w};
            v16h A;
            #pragma unroll
            for (int e = 0; e < 16; ++e) {
                const float p = (q == 0) ? ax[e]
                              : (q == 1) ? ay[e]
                              : (q == 2) ? ax[e] * ax[e]
                              : (q == 3) ? ay[e] * ay[e]
                              :            ax[e] * ay[e];
                A[e] = (_Float16)p;
            }
            v8f Z = {};
            // mid(16x16) = In(16x32) x W(32x16): f32 accumulate
            Dh[h] = __builtin_amdgcn_wmma_f32_16x16x32_f16(
                        false, A, false, W, (short)0, Z, false, false);
        }
        // D (f32) -> B (f16) operand conversion is purely per-lane (layouts share col->lane map)
        v16h Bv;
        #pragma unroll
        for (int e = 0; e < 8; ++e) {
            Bv[e]     = (_Float16)Dh[0][e];
            Bv[e + 8] = (_Float16)Dh[1][e];
        }
        v8f Z = {};
        // out(16x16) = W(16x32) x mid(32x16); mid rows 26..31 hit zero weight columns
        S[q] = __builtin_amdgcn_wmma_f32_16x16x32_f16(
                   false, W, false, Bv, (short)0, Z, false, false);
    }

    // ---- SSIM pointwise math (f32) + coalesced stores per D layout ----
    const int rbase = r0 + ((lane & 16) ? 8 : 0);
    const int col   = c0 + m;
    const float C1 = 1e-4f, C2 = 9e-4f;
    #pragma unroll
    for (int e = 0; e < 8; ++e) {
        const float mu1 = S[0][e], mu2 = S[1][e];
        const float ex2 = S[2][e], ey2 = S[3][e], exy = S[4][e];
        const float mu1s = mu1 * mu1, mu2s = mu2 * mu2, mu12 = mu1 * mu2;
        const float sg1 = ex2 - mu1s, sg2 = ey2 - mu2s, sg12 = exy - mu12;
        const float num  = (2.f * mu12 + C1) * (2.f * sg12 + C2);
        const float den  = (mu1s + mu2s + C1) * (sg1 + sg2 + C2);
        const float ssim = num / den;
        const float loss = fminf(fmaxf(1.f - ssim, 0.f), 1.f) * 0.5f;
        out[pbase + (size_t)(rbase + e) * HW + col] = loss;
    }
}

extern "C" void kernel_launch(void* const* d_in, const int* in_sizes, int n_in,
                              void* d_out, int out_size, void* d_ws, size_t ws_size,
                              hipStream_t stream) {
    (void)in_sizes; (void)n_in; (void)out_size; (void)d_ws; (void)ws_size;
    const float* img1 = (const float*)d_in[0];
    const float* img2 = (const float*)d_in[1];
    float* out = (float*)d_out;

    // 48 planes * (32x32) 16x16-tiles = 49152 tiles; 1 wave/tile, 4 waves/block
    const int numTiles = 16 * 3 * (HW / 16) * (HW / 16);
    const int blocks   = numTiles / WAVES_PER_BLOCK;   // 12288
    ssim_wmma_kernel<<<blocks, WAVES_PER_BLOCK * 32, 0, stream>>>(img1, img2, out);
}
